// TMDIntegrationLayer_50835232916409
// MI455X (gfx1250) — matrix operand order repaired and checked by reading, not compiled, not run on hardware
//
#include <hip/hip_runtime.h>

typedef __attribute__((ext_vector_type(16))) _Float16 v16h;
typedef __attribute__((ext_vector_type(8)))  float    v8f;
typedef __attribute__((ext_vector_type(2)))  _Float16 h2;

#define HIDDEN    64
#define K_BINS    10
#define PHI_BINS  10
#define N2_SLOTS  112   // 7 tiles * 16 (100 valid n2 points, padded)
#define TOT_SLOTS 128   // + 1 tile of 16 for n1 (10 valid)
#define NB        8     // batch elements per workgroup (amortize staging)
#define DK   (10.0f / 9.0f)
#define DPHI (3.14159265358979323846f / 9.0f)

__global__ __launch_bounds__(256) void tmd_fused_kernel(
    const float* __restrict__ qT,  const float* __restrict__ x1, const float* __restrict__ x2,
    const float* __restrict__ W1_1, const float* __restrict__ b1_1,
    const float* __restrict__ W2_1, const float* __restrict__ b2_1,
    const float* __restrict__ W3_1, const float* __restrict__ b3_1,
    const float* __restrict__ W1_2, const float* __restrict__ b1_2,
    const float* __restrict__ W2_2, const float* __restrict__ b2_2,
    const float* __restrict__ W3_2, const float* __restrict__ b3_2,
    float* __restrict__ out)
{
    // [net][n*64 + k]  (transposed so B-fragment K pairs are contiguous)
    __shared__ _Float16 sW2T[2][HIDDEN * HIDDEN];   // 16 KB
    __shared__ float    sH2[TOT_SLOTS][HIDDEN];     // 32 KB
    __shared__ _Float16 sW1h[2][2 * HIDDEN];        // W1 as f16: [0..63]=row0, [64..127]=row1
    __shared__ _Float16 sB1h[2][HIDDEN];
    __shared__ float    sB2[2][HIDDEN];
    __shared__ float    sW3[2][HIDDEN];
    __shared__ float    sB3[2];
    __shared__ float    sN1[16];
    __shared__ float    sVal[N2_SLOTS];
    __shared__ float    sContrib[TOT_SLOTS];

    const int tid  = threadIdx.x;
    const int lane = tid & 31;
    const int wave = tid >> 5;
    const int grp  = lane >> 4;     // half-wave group
    const int col  = lane & 15;

    // ---------------- stage weights into LDS (once per block) ----------------
    for (int i = tid; i < HIDDEN * HIDDEN; i += 256) {
        int k = i >> 6, n = i & 63;
        sW2T[0][n * HIDDEN + k] = (_Float16)W2_1[i];
        sW2T[1][n * HIDDEN + k] = (_Float16)W2_2[i];
    }
    for (int i = tid; i < 2 * HIDDEN; i += 256) {
        sW1h[0][i] = (_Float16)W1_1[i];  sW1h[1][i] = (_Float16)W1_2[i];
    }
    for (int i = tid; i < HIDDEN; i += 256) {
        sB1h[0][i] = (_Float16)b1_1[i];  sB1h[1][i] = (_Float16)b1_2[i];
        sB2[0][i]  = b2_1[i];            sB2[1][i]  = b2_2[i];
        sW3[0][i]  = W3_1[i];            sW3[1][i]  = W3_2[i];
    }
    if (tid == 0) { sB3[0] = b3_1[0]; sB3[1] = b3_2[0]; }
    __syncthreads();

    const int net = (wave == 7) ? 0 : 1;    // wave-uniform weight-set select

    // -------- B fragments + biases into registers (once per block) --------
    // B 32x16 f16 layout: lane holds N = t*16+col, K = grp*16 + 2v + s (+32 for kstep1)
    union BF { v16h v; unsigned u[8]; };
    BF bf0[4], bf1[4];
    float bias[4];
    #pragma unroll
    for (int t = 0; t < 4; ++t) {
        const unsigned* w2row = (const unsigned*)&sW2T[net][(t * 16 + col) * HIDDEN];
        #pragma unroll
        for (int r = 0; r < 8; ++r) {
            int kp = grp * 8 + r;           // dword index within 32-dword row
            bf0[t].u[r] = w2row[kp];
            bf1[t].u[r] = w2row[16 + kp];   // K += 32
        }
        bias[t] = sB2[net][t * 16 + col];
    }

    // -------- layer-1 weight pairs into registers (once per block) --------
    // A 16x32 f16 layout: lane row M = tile*16+col, K = grp*16 + 2*r + s
    const _Float16* w1net = sW1h[net];
    const _Float16* b1net = sB1h[net];
    h2 w0a[8], w1a[8], bba[8];   // for a0 (K = kb..kb+1)
    h2 w0b[8], w1b[8], bbb[8];   // for a1 (K += 32)
    #pragma unroll
    for (int r = 0; r < 8; ++r) {
        int kb = grp * 16 + (r << 1);   // even -> dword aligned
        w0a[r] = *(const h2*)&w1net[kb];
        w1a[r] = *(const h2*)&w1net[HIDDEN + kb];
        bba[r] = *(const h2*)&b1net[kb];
        w0b[r] = *(const h2*)&w1net[32 + kb];
        w1b[r] = *(const h2*)&w1net[HIDDEN + 32 + kb];
        bbb[r] = *(const h2*)&b1net[32 + kb];
    }

    // -------- per-lane geometry constants (batch-invariant) --------
    bool  valid;
    float c1 = 0.0f, c2 = 0.0f, in1fix = 0.0f;
    if (wave < 7) {                         // n2 point: kb = sqrt(q^2 + kv^2 - 2 q kv cos(phi))
        int p  = wave * 16 + col;
        valid  = p < (K_BINS * PHI_BINS);
        int ki = p / PHI_BINS, pj = p % PHI_BINS;
        float kv = (float)ki * DK;
        c1 = kv * kv;
        c2 = 2.0f * kv * cosf((float)pj * DPHI);
    } else {                                // n1 point: in = (x1, k)
        valid  = col < K_BINS;
        in1fix = (float)col * DK;
    }

    const int b0 = blockIdx.x * NB;
    const h2 zero2 = { (_Float16)0.0f, (_Float16)0.0f };

    for (int it = 0; it < NB; ++it) {
        const int b = b0 + it;
        const float q   = qT[b];
        const float x1v = x1[b];
        const float x2v = x2[b];

        float in0, in1;
        if (wave < 7) { in0 = x2v; in1 = sqrtf(fmaxf(q * q + c1 - q * c2, 0.0f)); }
        else          { in0 = x1v; in1 = in1fix; }

        // ---- layer 1 (2 -> 64) in packed f16, directly into A fragments ----
        const h2 vi0 = { (_Float16)in0, (_Float16)in0 };
        const h2 vi1 = { (_Float16)in1, (_Float16)in1 };
        union { v16h v; unsigned u[8]; } a0, a1;
        #pragma unroll
        for (int r = 0; r < 8; ++r) {
            union { h2 h; unsigned u; } t0, t1;
            t0.h = __builtin_elementwise_fma(vi0, w0a[r],
                   __builtin_elementwise_fma(vi1, w1a[r], bba[r]));
            t1.h = __builtin_elementwise_fma(vi0, w0b[r],
                   __builtin_elementwise_fma(vi1, w1b[r], bbb[r]));
            t0.h = __builtin_elementwise_max(t0.h, zero2);   // ReLU (packed)
            t1.h = __builtin_elementwise_max(t1.h, zero2);
            a0.u[r] = valid ? t0.u : 0u;                     // zero pad rows
            a1.u[r] = valid ? t1.u : 0u;
        }

        // ---- layer 2 (64 -> 64): 4 n-tiles x 2 k-steps of v_wmma_f32_16x16x32_f16 ----
        #pragma unroll
        for (int t = 0; t < 4; ++t) {
            const float bv = bias[t];
            v8f acc = { bv, bv, bv, bv, bv, bv, bv, bv };    // C[m][n] = b2[n]
            acc = __builtin_amdgcn_wmma_f32_16x16x32_f16(false, a0.v, false, bf0[t].v,
                                                         (short)0, acc, false, false);
            acc = __builtin_amdgcn_wmma_f32_16x16x32_f16(false, a1.v, false, bf1[t].v,
                                                         (short)0, acc, false, false);
            // ReLU + scatter D (C/D layout: M = r + 8*grp, N = t*16+col).
            // NOTE: keep this as compiler-visible fmaxf so the backend inserts the
            // WMMA->VALU RAW hazard NOPs/delays (inline asm consumers bypass them).
            #pragma unroll
            for (int r = 0; r < 8; ++r) {
                sH2[wave * 16 + r + 8 * grp][t * 16 + col] = fmaxf(acc[r], 0.0f);
            }
        }
        __syncthreads();

        // ---- layer 3 (64 -> 1): 2 threads per row, all 256 threads busy ----
        {
            const int slot = tid >> 1;
            const int hf   = tid & 1;
            const int ln   = (slot < N2_SLOTS) ? 1 : 0;
            float acc = 0.0f;
            #pragma unroll
            for (int j = 0; j < 32; ++j) {
                int jj = hf * 32 + j;
                acc += sH2[slot][jj] * sW3[ln][jj];
            }
            acc += __shfl_down(acc, 1, 32);      // combine the two halves (adjacent lanes)
            if (hf == 0) {
                float v = acc + sB3[ln];
                if (slot < N2_SLOTS) {
                    sVal[slot] = v;
                } else {
                    int ks = slot - N2_SLOTS;
                    if (ks < K_BINS) sN1[ks] = v;
                }
            }
        }
        __syncthreads();

        // ---- contrib = n1[k] * n2 * k_vals[k] * dk*dphi ; reduce ----
        if (tid < TOT_SLOTS) {
            float c = 0.0f;
            if (tid < K_BINS * PHI_BINS) {
                int ki = tid / PHI_BINS;
                c = sN1[ki] * sVal[tid] * ((float)ki * DK) * (DK * DPHI);
            }
            sContrib[tid] = c;
        }
        __syncthreads();
        if (tid < 32) {
            float s = sContrib[tid] + sContrib[tid + 32] + sContrib[tid + 64] + sContrib[tid + 96];
            #pragma unroll
            for (int off = 16; off > 0; off >>= 1)
                s += __shfl_down(s, off, 32);
            if (tid == 0) out[b] = s;
        }
        // no trailing barrier needed: next-iter sH2/sVal/sContrib writes are ordered
        // behind this iteration's barriers for every producer/consumer pair
    }
}

extern "C" void kernel_launch(void* const* d_in, const int* in_sizes, int n_in,
                              void* d_out, int out_size, void* d_ws, size_t ws_size,
                              hipStream_t stream) {
    (void)n_in; (void)out_size; (void)d_ws; (void)ws_size;
    const float* qT   = (const float*)d_in[0];
    const float* x1   = (const float*)d_in[1];
    const float* x2   = (const float*)d_in[2];
    const float* W1_1 = (const float*)d_in[3];
    const float* b1_1 = (const float*)d_in[4];
    const float* W2_1 = (const float*)d_in[5];
    const float* b2_1 = (const float*)d_in[6];
    const float* W3_1 = (const float*)d_in[7];
    const float* b3_1 = (const float*)d_in[8];
    const float* W1_2 = (const float*)d_in[9];
    const float* b1_2 = (const float*)d_in[10];
    const float* W2_2 = (const float*)d_in[11];
    const float* b2_2 = (const float*)d_in[12];
    const float* W3_2 = (const float*)d_in[13];
    const float* b3_2 = (const float*)d_in[14];
    float* out = (float*)d_out;

    const int batch = in_sizes[0];           // 8192
    const int grid  = (batch + NB - 1) / NB; // 1024 workgroups
    tmd_fused_kernel<<<grid, 256, 0, stream>>>(
        qT, x1, x2,
        W1_1, b1_1, W2_1, b2_1, W3_1, b3_1,
        W1_2, b1_2, W2_2, b2_2, W3_2, b3_2,
        out);
}
